// SNNModel_46445776339008
// MI455X (gfx1250) — matrix operand order repaired and checked by reading, not compiled, not run on hardware
//
#include <hip/hip_runtime.h>
#include <hip/hip_bf16.h>

typedef __attribute__((ext_vector_type(16))) _Float16 v16h;
typedef __attribute__((ext_vector_type(8)))  _Float16 v8h;
typedef __attribute__((ext_vector_type(8)))  float    v8f;

#define D_IN   512
#define H0     1024
#define H1     512
#define M_TILE 32            // rows per workgroup (2 WMMA M sub-tiles)
#define NWAVES 16            // 512 threads
#define SPD    (D_IN + 8)    // padded row stride (halves) for staged data hi/lo
#define SP1    (H0 + 8)      // padded row stride (halves) for spk1
#define RAW_BYTES   (M_TILE * SP1 * 2)                  // 66048 B: spk1 region, aliases raw f32 tile (65536 B)
#define HILO_BYTES  (M_TILE * SPD * 2)                  // 33280 B each
#define SMEM_BYTES  (RAW_BYTES + 2 * HILO_BYTES + M_TILE * (int)sizeof(float))

__device__ __forceinline__ v8f wmma_f16(v16h a, v16h b, v8f c) {
    return __builtin_amdgcn_wmma_f32_16x16x32_f16(
        /*neg_a=*/false, a, /*neg_b=*/false, b,
        /*c_mod=*/(short)0, c, /*reuse_a=*/false, /*reuse_b=*/false);
}

// A-fragment (16x32 f16, MxK) from LDS, row stride in halves.
// Contiguous k order (same as B): lanes 0-15 -> M=lane, K 0..15; lanes 16-31 -> K 16..31
__device__ __forceinline__ v16h load_a_lds(const _Float16* s, int rowStride,
                                           int k0, int lane) {
    int row = lane & 15;
    int kb  = (lane < 16) ? 0 : 16;
    const _Float16* p = s + row * rowStride + k0 + kb;
    v8h a0 = *(const v8h*)(p);        // K = k0+kb+0..7
    v8h a1 = *(const v8h*)(p + 8);    // K = k0+kb+8..15
    return __builtin_shufflevector(a0, a1, 0, 1, 2, 3, 4, 5, 6, 7,
                                           8, 9, 10, 11, 12, 13, 14, 15);
}

// B-fragment (32x16 f16, KxN) from pre-split row-major f16 weight [N][ldw].
// Contiguous k order (same as A): lanes 0-15 -> N=lane, K=0..15; lanes 16-31 -> N=lane-16, K=16..31
__device__ __forceinline__ v16h load_b_f16(const _Float16* __restrict__ w, int ldw,
                                           int nbase, int k0, int lane) {
    int n  = nbase + (lane & 15);
    int kb = (lane < 16) ? 0 : 16;
    return *(const v16h*)(w + (size_t)n * ldw + k0 + kb);   // 32B aligned
}

// ---- prep: split w1/w2 into hi/lo f16 pairs (once per launch, L2-resident) ----
__global__ void split_weights_kernel(const float* __restrict__ w1,
                                     const float* __restrict__ w2,
                                     _Float16* __restrict__ w1h,
                                     _Float16* __restrict__ w1l,
                                     _Float16* __restrict__ w2h,
                                     _Float16* __restrict__ w2l) {
    const int n1 = H0 * D_IN;
    const int n2 = H1 * H0;
    for (int i = blockIdx.x * blockDim.x + threadIdx.x; i < n1 + n2;
         i += gridDim.x * blockDim.x) {
        if (i < n1) {
            float f = w1[i];
            _Float16 h = (_Float16)f;
            w1h[i] = h;
            w1l[i] = (_Float16)(f - (float)h);
        } else {
            int j = i - n1;
            float f = w2[j];
            _Float16 h = (_Float16)f;
            w2h[j] = h;
            w2l[j] = (_Float16)(f - (float)h);
        }
    }
}

__global__ void snn_fused_kernel(const float* __restrict__ data,
                                 const _Float16* __restrict__ w1h,
                                 const _Float16* __restrict__ w1l,
                                 const _Float16* __restrict__ w2h,
                                 const _Float16* __restrict__ w2l,
                                 const float* __restrict__ b1,
                                 const float* __restrict__ b2,
                                 const float* __restrict__ w3,
                                 const float* __restrict__ b3,
                                 const int*   __restrict__ tsteps,
                                 float* __restrict__ out) {
    extern __shared__ char smem[];
    float*    sRaw = (float*)smem;                    // 32x512 f32 (staging, dead after convert)
    _Float16* sSpk = (_Float16*)smem;                 // 32*SP1 halves (aliases sRaw)
    _Float16* sHi  = (_Float16*)(smem + RAW_BYTES);   // 32*SPD halves
    _Float16* sLo  = sHi + M_TILE * SPD;              // 32*SPD halves
    float*    sC3  = (float*)(smem + RAW_BYTES + 2 * HILO_BYTES);

    const int tid   = threadIdx.x;
    const int lane  = tid & 31;
    const int wave  = tid >> 5;
    const int rbase = (lane < 16) ? 0 : 8;            // C/D: M = reg + rbase (+16*m)
    const int T     = tsteps[0];

    if (tid < M_TILE) sC3[tid] = 0.0f;

    // ---- stage raw data tile [32 x 512] f32 into LDS via gfx1250 async copy ----
    {
        const unsigned long long gbase =
            (unsigned long long)(data + (size_t)blockIdx.x * M_TILE * D_IN);
#pragma unroll
        for (int q = 0; q < 8; ++q) {
            int idx    = tid + q * (NWAVES * 32);     // float4 index 0..4095
            int byteOf = idx * 16;                    // same offset in LDS and global
            asm volatile("global_load_async_to_lds_b128 %0, %1, %2 offset:0"
                         :: "v"(byteOf), "v"(byteOf), "s"(gbase)
                         : "memory");
        }
        asm volatile("s_wait_asynccnt 0" ::: "memory");
    }
    __syncthreads();

    // ---- convert staged tile to hi/lo f16 split ----
    for (int idx = tid; idx < M_TILE * D_IN; idx += blockDim.x) {
        int row = idx >> 9;                           // /512
        int c   = idx & (D_IN - 1);
        float f = sRaw[idx];
        _Float16 h = (_Float16)f;
        sHi[row * SPD + c] = h;
        sLo[row * SPD + c] = (_Float16)(f - (float)h);
    }
    __syncthreads();

    // ---- phase 1: cur1 = data @ w1^T + b1, register resident ----
    // per wave: 4 N-tiles x 2 M sub-tiles; each B fragment reused across both M tiles
    v8f cur1[2][4];
#pragma unroll
    for (int m = 0; m < 2; ++m)
#pragma unroll
        for (int i = 0; i < 4; ++i) cur1[m][i] = (v8f)(0.0f);

    for (int k0 = 0; k0 < D_IN; k0 += 32) {
        v16h ah[2], al[2];
#pragma unroll
        for (int m = 0; m < 2; ++m) {
            ah[m] = load_a_lds(sHi + m * 16 * SPD, SPD, k0, lane);
            al[m] = load_a_lds(sLo + m * 16 * SPD, SPD, k0, lane);
        }
#pragma unroll
        for (int i = 0; i < 4; ++i) {
            int nbase = (wave + NWAVES * i) * 16;
            v16h bh = load_b_f16(w1h, D_IN, nbase, k0, lane);
            v16h bl = load_b_f16(w1l, D_IN, nbase, k0, lane);
#pragma unroll
            for (int m = 0; m < 2; ++m) {
                cur1[m][i] = wmma_f16(ah[m], bh, cur1[m][i]);
                cur1[m][i] = wmma_f16(ah[m], bl, cur1[m][i]);
                cur1[m][i] = wmma_f16(al[m], bh, cur1[m][i]);
            }
        }
    }
#pragma unroll
    for (int i = 0; i < 4; ++i) {
        float bb = b1[(wave + NWAVES * i) * 16 + (lane & 15)];
#pragma unroll
        for (int m = 0; m < 2; ++m)
#pragma unroll
            for (int r = 0; r < 8; ++r) cur1[m][i][r] += bb;
    }

    // loop-invariant layer-2 per-column constants
    float bb2[2], w3c[2];
#pragma unroll
    for (int i = 0; i < 2; ++i) {
        int col = (wave + NWAVES * i) * 16 + (lane & 15);
        bb2[i]  = b2[col];
        w3c[i]  = w3[col];
    }
    __syncthreads();   // done with sRaw/sHi/sLo readers; sSpk may now be written

    // ---- temporal recurrence ----
    v8f mem1[2][4], mem2[2][2];
#pragma unroll
    for (int m = 0; m < 2; ++m) {
#pragma unroll
        for (int i = 0; i < 4; ++i) mem1[m][i] = (v8f)(0.0f);
#pragma unroll
        for (int i = 0; i < 2; ++i) mem2[m][i] = (v8f)(0.0f);
    }

    for (int t = 0; t < T; ++t) {
        // LIF layer 1 (beta1=1.0, reset detected on OLD mem) -> spk1 to LDS (f16 exact)
#pragma unroll
        for (int m = 0; m < 2; ++m)
#pragma unroll
            for (int i = 0; i < 4; ++i) {
                int col = (wave + NWAVES * i) * 16 + (lane & 15);
#pragma unroll
                for (int r = 0; r < 8; ++r) {
                    float mv    = mem1[m][i][r];
                    float reset = (mv > 1.0f) ? 1.0f : 0.0f;
                    mv          = mv + cur1[m][i][r] - reset;
                    mem1[m][i][r] = mv;
                    float spk   = (mv - 1.0f > 0.0f) ? 1.0f : 0.0f;
                    sSpk[(m * 16 + rbase + r) * SP1 + col] = (_Float16)spk;
                }
            }
        __syncthreads();

        // cur2 = spk1 @ w2^T  (A exact f16, B pre-split hi/lo, reused across M tiles)
        v8f acc[2][2];
#pragma unroll
        for (int m = 0; m < 2; ++m)
#pragma unroll
            for (int i = 0; i < 2; ++i) acc[m][i] = (v8f)(0.0f);
        for (int k0 = 0; k0 < H0; k0 += 32) {
            v16h a[2];
#pragma unroll
            for (int m = 0; m < 2; ++m)
                a[m] = load_a_lds(sSpk + m * 16 * SP1, SP1, k0, lane);
#pragma unroll
            for (int i = 0; i < 2; ++i) {
                int nbase = (wave + NWAVES * i) * 16;
                const _Float16* ph = w2h + (size_t)(nbase + (lane & 15)) * H0 + k0;
                __builtin_prefetch(ph + 32, 0, 1);            // next K-step, hi stream
                v16h bh = load_b_f16(w2h, H0, nbase, k0, lane);
                v16h bl = load_b_f16(w2l, H0, nbase, k0, lane);
#pragma unroll
                for (int m = 0; m < 2; ++m) {
                    acc[m][i] = wmma_f16(a[m], bh, acc[m][i]);
                    acc[m][i] = wmma_f16(a[m], bl, acc[m][i]);
                }
            }
        }

        // LIF layer 2 (beta2=0.95) + cur3 partial dot with w3
        float part[2][8];
#pragma unroll
        for (int m = 0; m < 2; ++m)
#pragma unroll
            for (int r = 0; r < 8; ++r) part[m][r] = 0.0f;
#pragma unroll
        for (int m = 0; m < 2; ++m)
#pragma unroll
            for (int i = 0; i < 2; ++i) {
#pragma unroll
                for (int r = 0; r < 8; ++r) {
                    float mv    = mem2[m][i][r];
                    float reset = (mv > 1.0f) ? 1.0f : 0.0f;
                    mv          = 0.95f * mv + (acc[m][i][r] + bb2[i]) - reset;
                    mem2[m][i][r] = mv;
                    float s2    = (mv - 1.0f > 0.0f) ? 1.0f : 0.0f;
                    part[m][r] += s2 * w3c[i];
                }
            }
#pragma unroll
        for (int m = 0; m < 2; ++m)
#pragma unroll
            for (int r = 0; r < 8; ++r)
                atomicAdd(&sC3[m * 16 + rbase + r], part[m][r]);
        __syncthreads();   // sC3 adds visible; safe to rewrite sSpk next step
    }

    if (tid < M_TILE) {
        out[(size_t)blockIdx.x * M_TILE + tid] = sC3[tid] / (float)T + b3[0];
    }
}

extern "C" void kernel_launch(void* const* d_in, const int* in_sizes, int n_in,
                              void* d_out, int out_size, void* d_ws, size_t ws_size,
                              hipStream_t stream) {
    const float* data = (const float*)d_in[0];
    const float* w1   = (const float*)d_in[1];
    const float* b1   = (const float*)d_in[2];
    const float* w2   = (const float*)d_in[3];
    const float* b2   = (const float*)d_in[4];
    const float* w3   = (const float*)d_in[5];
    const float* b3   = (const float*)d_in[6];
    const int*   ts   = (const int*)d_in[7];
    float*       out  = (float*)d_out;

    // workspace layout: pre-split f16 weights (4 MB total, L2-resident)
    _Float16* w1h = (_Float16*)d_ws;
    _Float16* w1l = w1h + (size_t)H0 * D_IN;
    _Float16* w2h = w1l + (size_t)H0 * D_IN;
    _Float16* w2l = w2h + (size_t)H1 * H0;

    split_weights_kernel<<<dim3(1024), dim3(256), 0, stream>>>(w1, w2, w1h, w1l, w2h, w2l);

    const int B = in_sizes[0] / D_IN;            // 16384
    dim3 grid(B / M_TILE);                       // 512 workgroups, 32 rows each
    dim3 block(NWAVES * 32);                     // 512 threads = 16 waves

    snn_fused_kernel<<<grid, block, (size_t)SMEM_BYTES, stream>>>(
        data, w1h, w1l, w2h, w2l, b1, b2, w3, b3, ts, out);
}
